// SWIFTModel_25666724560958
// MI455X (gfx1250) — compile-verified, matrix operand order checked by reading, not run on hardware
//
#include <hip/hip_runtime.h>

#define DEV_INLINE __device__ __forceinline__

typedef __bf16 bf16;
typedef __attribute__((ext_vector_type(16))) __bf16 v16bf;
typedef __attribute__((ext_vector_type(8)))  float  v8f;

// ---- geometry (fixed by the reference problem) ----
static constexpr int BATCH = 32;
static constexpr int LSEQ  = 1024;   // spatial sequence length (== 2*MN)
static constexpr int MN    = 512;    // velocity sequence length
static constexpr int SD    = 256;    // spatial_dim
static constexpr int VD    = 64;     // velocity_dim
static constexpr int DM    = 256;    // d_model
static constexpr int D3    = 192;    // 3*velocity_dim
static constexpr int NH    = 8;      // heads
static constexpr int DK    = 32;     // head dim
static constexpr int MAXR  = 30;     // max relative position

// -------------------------------------------------------------------------
// WMMA fragment loader: A/B operand layout for v_wmma_*_16x16x32_bf16.
// Per lane: 8 contiguous bf16 at K=kb..kb+7 and 8 at K=kb+16..kb+23.
// Caller passes p = &tile[row*32 + kb].
// -------------------------------------------------------------------------
DEV_INLINE v16bf load_frag16(const bf16* p) {
  v16bf r;
#pragma unroll
  for (int i = 0; i < 8; ++i) { r[i] = p[i]; r[i + 8] = p[i + 16]; }
  return r;
}

// -------------------------------------------------------------------------
// VALU 16-lane reductions via v_permlane16_b32 (no LDS / ds_bpermute).
// Lane-select nibble tables implement xor-1/2/4/8 permutations; each is an
// involution so the network fully reduces within each 16-lane half.
// -------------------------------------------------------------------------
DEV_INLINE float permlane16_f(float x, unsigned lo, unsigned hi) {
  int xi = __float_as_int(x);
  int r = __builtin_amdgcn_permlane16(xi, xi, (int)lo, (int)hi, false, false);
  return __int_as_float(r);
}
DEV_INLINE float redmax16(float x) {
  x = fmaxf(x, permlane16_f(x, 0x67452301u, 0xEFCDAB89u));  // xor 1
  x = fmaxf(x, permlane16_f(x, 0x54761032u, 0xDCFE98BAu));  // xor 2
  x = fmaxf(x, permlane16_f(x, 0x32107654u, 0xBA98FEDCu));  // xor 4
  x = fmaxf(x, permlane16_f(x, 0xFEDCBA98u, 0x76543210u));  // xor 8
  return x;
}
DEV_INLINE float redsum16(float x) {
  x += permlane16_f(x, 0x67452301u, 0xEFCDAB89u);
  x += permlane16_f(x, 0x54761032u, 0xDCFE98BAu);
  x += permlane16_f(x, 0x32107654u, 0xBA98FEDCu);
  x += permlane16_f(x, 0xFEDCBA98u, 0x76543210u);
  return x;
}

// -------------------------------------------------------------------------
// f32 -> bf16 conversion (weights)
// -------------------------------------------------------------------------
__global__ void cvt_f32_bf16_kernel(const float* __restrict__ s,
                                    bf16* __restrict__ d, int n) {
  int i = blockIdx.x * blockDim.x + threadIdx.x;
  if (i < n) d[i] = (bf16)s[i];
}

// -------------------------------------------------------------------------
// Multi-scale 1D convs (k=3,5,7, 'same'), concat on channel -> xm [B,MN,192]
// One block per (b,j). 192 threads: thread c computes output channel c.
// -------------------------------------------------------------------------
__global__ __launch_bounds__(192)
void msconv_kernel(const float* __restrict__ xv,
                   const float* __restrict__ w3, const float* __restrict__ b3,
                   const float* __restrict__ w5, const float* __restrict__ b5,
                   const float* __restrict__ w7, const float* __restrict__ b7,
                   float* __restrict__ xm) {
  const int b = blockIdx.x / MN;
  const int j = blockIdx.x % MN;
  __shared__ float sx[7 * VD];   // rows j-3 .. j+3, zero padded
  for (int idx = threadIdx.x; idx < 7 * VD; idx += 192) {
    int r  = idx / VD - 3;
    int ci = idx % VD;
    int jj = j + r;
    sx[idx] = (jj >= 0 && jj < MN) ? xv[((size_t)b * MN + jj) * VD + ci] : 0.0f;
  }
  __syncthreads();
  const int c = threadIdx.x;
  const float* wp; int K; int co; float acc;
  if (c < 64)       { wp = w3; K = 3; co = c;       acc = b3[co]; }
  else if (c < 128) { wp = w5; K = 5; co = c - 64;  acc = b5[co]; }
  else              { wp = w7; K = 7; co = c - 128; acc = b7[co]; }
  const int r0 = 3 - (K >> 1);
  for (int k = 0; k < K; ++k) {
    const float* xrow = &sx[(r0 + k) * VD];
    const float* wrow = &wp[((size_t)co * VD) * K + k];
#pragma unroll 8
    for (int ci = 0; ci < VD; ++ci) acc += xrow[ci] * wrow[(size_t)ci * K];
  }
  xm[((size_t)b * MN + j) * D3 + c] = acc;
}

// -------------------------------------------------------------------------
// pos_sum[j,c] = sum_i emb[clip(j-i,-R,R)+R, c]  (batch independent)
// -------------------------------------------------------------------------
__global__ __launch_bounds__(192)
void possum_kernel(const float* __restrict__ emb, float* __restrict__ ps) {
  const int j = blockIdx.x;
  const int c = threadIdx.x;
  const int lo = j - (MN - 1), hi = j;
  const int a = lo > -MAXR ? lo : -MAXR;
  const int bnd = hi < MAXR ? hi : MAXR;
  float sum = 0.0f;
  for (int d = a; d <= bnd; ++d) sum += emb[(size_t)(d + MAXR) * D3 + c];
  if (lo < -MAXR) sum += (float)(-MAXR - lo) * emb[c];
  if (hi >  MAXR) sum += (float)(hi - MAXR) * emb[(size_t)(2 * MAXR) * D3 + c];
  ps[(size_t)j * D3 + c] = sum;
}

// -------------------------------------------------------------------------
// Relative positional encoding closed form -> x_v_pe (bf16) [B,MN,192]
// -------------------------------------------------------------------------
__global__ __launch_bounds__(192)
void relpe_kernel(const float* __restrict__ xm, const float* __restrict__ emb,
                  const float* __restrict__ dww, const float* __restrict__ dwb,
                  const float* __restrict__ ps, bf16* __restrict__ xvpe) {
  const int b = blockIdx.x / MN;
  const int j = blockIdx.x % MN;
  const int c = threadIdx.x;
  const size_t base = ((size_t)b * MN + j) * D3 + c;
  const float xc = xm[base];
  const float fC = xc + emb[(size_t)MAXR * D3 + c];
  const float fL = (j > 0)      ? xm[base - D3] + emb[(size_t)(MAXR - 1) * D3 + c] : 0.0f;
  const float fR = (j < MN - 1) ? xm[base + D3] + emb[(size_t)(MAXR + 1) * D3 + c] : 0.0f;
  const float dc = fL * dww[c * 3 + 0] + fC * dww[c * 3 + 1] + fR * dww[c * 3 + 2] + dwb[c];
  const float v  = xc + (ps[(size_t)j * D3 + c] - fC + dc) * (1.0f / (float)MN);
  xvpe[base] = (bf16)v;
}

// -------------------------------------------------------------------------
// Generic WMMA GEMM:  C[M,N] = act( scale * (A[M,KD] @ B[N,KD]^T) + bias )
// BM=128, BN=64, K-step 32, 256 threads (8 waves), wave w -> rows w*16..+16.
// -------------------------------------------------------------------------
template <int KD, bool A_F32, bool OUT_BF16, bool SIGMOID>
__global__ __launch_bounds__(256)
void gemm_wmma_kernel(const void* __restrict__ Av, const bf16* __restrict__ B,
                      const float* __restrict__ bias, void* __restrict__ Cv,
                      int N, float scale) {
  __shared__ bf16 sA[128 * 32];
  __shared__ bf16 sB[64 * 32];

  const int tid  = threadIdx.x;
  const int lane = tid & 31;
  const int wave = tid >> 5;
  const int half = lane >> 4;
  const int mrow = lane & 15;
  const int kb   = half * 8;
  const int blockN = blockIdx.x * 64;
  const int blockM = blockIdx.y * 128;

  v8f acc[4];
#pragma unroll
  for (int c = 0; c < 4; ++c) acc[c] = (v8f){0.f, 0.f, 0.f, 0.f, 0.f, 0.f, 0.f, 0.f};

  for (int k0 = 0; k0 < KD; k0 += 32) {
    __syncthreads();
    {   // stage A tile 128x32 (convert f32->bf16 if needed)
      const int base = tid * 16, row = base >> 5, col = base & 31;
      const size_t g = (size_t)(blockM + row) * KD + k0 + col;
      if (A_F32) {
        const float* Af = (const float*)Av + g;
#pragma unroll
        for (int i = 0; i < 16; ++i) sA[row * 32 + col + i] = (bf16)Af[i];
      } else {
        const bf16* Ab = (const bf16*)Av + g;
#pragma unroll
        for (int i = 0; i < 16; ++i) sA[row * 32 + col + i] = Ab[i];
      }
    }
    {   // stage B tile 64x32
      const int base = tid * 8, row = base >> 5, col = base & 31;
      const bf16* Bp = B + (size_t)(blockN + row) * KD + k0 + col;
#pragma unroll
      for (int i = 0; i < 8; ++i) sB[row * 32 + col + i] = Bp[i];
    }
    __syncthreads();

    const v16bf a = load_frag16(&sA[(wave * 16 + mrow) * 32 + kb]);
#pragma unroll
    for (int c = 0; c < 4; ++c) {
      const v16bf bb = load_frag16(&sB[(c * 16 + mrow) * 32 + kb]);
      acc[c] = __builtin_amdgcn_wmma_f32_16x16x32_bf16(
          false, a, false, bb, (short)0, acc[c], false, false);
    }
  }

#pragma unroll
  for (int c = 0; c < 4; ++c) {
#pragma unroll
    for (int r = 0; r < 8; ++r) {
      const int m = wave * 16 + r + half * 8;
      const int n = blockN + c * 16 + mrow;
      float v = acc[c][r] * scale + (bias ? bias[n] : 0.0f);
      if (SIGMOID) v = 1.0f / (1.0f + __expf(-v));
      const size_t idx = (size_t)(blockM + m) * N + n;
      if (OUT_BF16) ((bf16*)Cv)[idx] = (bf16)v;
      else          ((float*)Cv)[idx] = v;
    }
  }
}

// -------------------------------------------------------------------------
// Flash-style attention: one block per (b, h, 128 Q rows). 8 waves; wave w
// owns Q rows w*16..+16 with dk=32 (two 16x16 output tiles). Keys streamed
// in chunks of 32 through LDS; online softmax (VALU permlane reductions);
// P re-shaped via per-wave LDS. Q pre-scaled by 1/sqrt(dk).
// -------------------------------------------------------------------------
__global__ __launch_bounds__(256)
void attn_kernel(const bf16* __restrict__ Q, const bf16* __restrict__ Km,
                 const bf16* __restrict__ Vm, bf16* __restrict__ O) {
  __shared__ bf16 sK[32 * 32];        // [key][feat]
  __shared__ bf16 sVt[32 * 32];       // [d][key]  (transposed V chunk)
  __shared__ bf16 sP[8 * 16 * 32];    // per-wave P tile [16][32]

  const int tid  = threadIdx.x;
  const int lane = tid & 31;
  const int wave = tid >> 5;
  const int half = lane >> 4;
  const int mrow = lane & 15;
  const int kb   = half * 8;

  const int b = blockIdx.z, h = blockIdx.y;
  const int qrow0 = blockIdx.x * 128;

  const bf16* Kb = Km + (size_t)b * MN * DM + h * DK;
  const bf16* Vb = Vm + (size_t)b * MN * DM + h * DK;

  // Q fragment (A operand), rows fixed for the whole kernel
  const bf16* Qp = Q + ((size_t)b * LSEQ + qrow0 + wave * 16 + mrow) * DM + h * DK + kb;
  const v16bf qfrag = load_frag16(Qp);

  v8f o0 = (v8f){0.f,0.f,0.f,0.f,0.f,0.f,0.f,0.f};
  v8f o1 = (v8f){0.f,0.f,0.f,0.f,0.f,0.f,0.f,0.f};
  float m_run[8], l_run[8];
#pragma unroll
  for (int r = 0; r < 8; ++r) { m_run[r] = -1e30f; l_run[r] = 0.0f; }

  bf16* sPw = &sP[wave * 16 * 32];

  for (int s0 = 0; s0 < MN; s0 += 32) {
    __syncthreads();  // protect previous-iteration LDS readers
    {   // stage K chunk [32 keys][32 feats] and V chunk transposed [32 d][32 keys]
      const int idx = tid * 4;
      const int krow = idx >> 5, kcol = idx & 31;
      const bf16* kp = Kb + (size_t)(s0 + krow) * DM + kcol;
      const bf16* vp = Vb + (size_t)(s0 + krow) * DM + kcol;
#pragma unroll
      for (int j = 0; j < 4; ++j) {
        sK[krow * 32 + kcol + j] = kp[j];
        sVt[(kcol + j) * 32 + krow] = vp[j];
      }
    }
    __syncthreads();

    // S = Q K^T : two 16x16 tiles (keys s0..+15 and s0+16..+31)
    const v16bf kb0 = load_frag16(&sK[mrow * 32 + kb]);
    const v16bf kb1 = load_frag16(&sK[(16 + mrow) * 32 + kb]);
    const v8f zero = (v8f){0.f,0.f,0.f,0.f,0.f,0.f,0.f,0.f};
    v8f sf0 = __builtin_amdgcn_wmma_f32_16x16x32_bf16(false, qfrag, false, kb0,
                                                      (short)0, zero, false, false);
    v8f sf1 = __builtin_amdgcn_wmma_f32_16x16x32_bf16(false, qfrag, false, kb1,
                                                      (short)0, zero, false, false);

    // online softmax per row; rows live in 16-lane halves of the wave;
    // cross-lane reductions stay in the VALU via v_permlane16_b32
#pragma unroll
    for (int r = 0; r < 8; ++r) {
      const float mx = redmax16(fmaxf(sf0[r], sf1[r]));
      const float mnew = fmaxf(m_run[r], mx);
      const float p0 = __expf(sf0[r] - mnew);
      const float p1 = __expf(sf1[r] - mnew);
      const float alpha = __expf(m_run[r] - mnew);
      const float rs = redsum16(p0 + p1);
      l_run[r] = l_run[r] * alpha + rs;
      m_run[r] = mnew;
      o0[r] *= alpha;
      o1[r] *= alpha;
      const int prow = r + half * 8;
      sPw[prow * 32 + mrow]      = (bf16)p0;
      sPw[prow * 32 + 16 + mrow] = (bf16)p1;
    }
    __syncthreads();  // P tiles visible (uniform barrier across all waves)

    // O += P V : A = P [16x32], B = V^T tiles
    const v16bf pa  = load_frag16(&sPw[mrow * 32 + kb]);
    const v16bf bv0 = load_frag16(&sVt[mrow * 32 + kb]);
    const v16bf bv1 = load_frag16(&sVt[(16 + mrow) * 32 + kb]);
    o0 = __builtin_amdgcn_wmma_f32_16x16x32_bf16(false, pa, false, bv0,
                                                 (short)0, o0, false, false);
    o1 = __builtin_amdgcn_wmma_f32_16x16x32_bf16(false, pa, false, bv1,
                                                 (short)0, o1, false, false);
  }

  // normalize and store (bf16, head-major channel layout)
#pragma unroll
  for (int r = 0; r < 8; ++r) {
    const float inv = 1.0f / l_run[r];
    const size_t grow = (size_t)b * LSEQ + qrow0 + wave * 16 + r + half * 8;
    O[grow * DM + h * DK + mrow]      = (bf16)(o0[r] * inv);
    O[grow * DM + h * DK + 16 + mrow] = (bf16)(o1[r] * inv);
  }
}

// -------------------------------------------------------------------------
// Final: out = LN( x_spatial + tile(Gamma) * fc_out )  (one block per row)
// -------------------------------------------------------------------------
__global__ __launch_bounds__(256)
void ln_final_kernel(const float* __restrict__ fc_out,
                     const float* __restrict__ gamma,
                     const float* __restrict__ x_spatial,
                     const float* __restrict__ ln_w,
                     const float* __restrict__ ln_b,
                     float* __restrict__ out) {
  __shared__ float rs[256], rq[256];
  const int row = blockIdx.x;               // b*LSEQ + l
  const int b = row / LSEQ, l = row % LSEQ;
  const int c = threadIdx.x;
  const float g = gamma[((size_t)b * MN + (l & (MN - 1))) * DM + c];
  const float v = fc_out[(size_t)row * DM + c] * g + x_spatial[(size_t)row * DM + c];
  rs[c] = v; rq[c] = v * v;
  __syncthreads();
  for (int st = 128; st > 0; st >>= 1) {
    if (c < st) { rs[c] += rs[c + st]; rq[c] += rq[c + st]; }
    __syncthreads();
  }
  const float mu  = rs[0] * (1.0f / DM);
  const float var = rq[0] * (1.0f / DM) - mu * mu;
  out[(size_t)row * DM + c] = (v - mu) * rsqrtf(var + 1e-5f) * ln_w[c] + ln_b[c];
}

// -------------------------------------------------------------------------
// Launcher
// -------------------------------------------------------------------------
extern "C" void kernel_launch(void* const* d_in, const int* in_sizes, int n_in,
                              void* d_out, int out_size, void* d_ws, size_t ws_size,
                              hipStream_t stream) {
  (void)in_sizes; (void)n_in; (void)out_size; (void)ws_size;
  const float* x_spatial  = (const float*)d_in[0];
  const float* x_velocity = (const float*)d_in[1];
  const float* Wg    = (const float*)d_in[2];
  const float* ms_w3 = (const float*)d_in[3];
  const float* ms_b3 = (const float*)d_in[4];
  const float* ms_w5 = (const float*)d_in[5];
  const float* ms_b5 = (const float*)d_in[6];
  const float* ms_w7 = (const float*)d_in[7];
  const float* ms_b7 = (const float*)d_in[8];
  const float* rel_emb = (const float*)d_in[9];
  const float* dw_w  = (const float*)d_in[10];
  const float* dw_b  = (const float*)d_in[11];
  const float* Wq    = (const float*)d_in[12];
  const float* Wk    = (const float*)d_in[13];
  const float* Wv    = (const float*)d_in[14];
  const float* fc_w  = (const float*)d_in[15];
  const float* fc_b  = (const float*)d_in[16];
  const float* ln_w  = (const float*)d_in[17];
  const float* ln_b  = (const float*)d_in[18];
  float* out = (float*)d_out;

  // workspace layout
  char* ws = (char*)d_ws;
  size_t off = 0;
  auto alloc = [&](size_t bytes) {
    size_t o = off; off = (off + bytes + 255) & ~(size_t)255; return o;
  };
  bf16*  wq_bf   = (bf16*)(ws + alloc((size_t)DM * SD * 2));
  bf16*  wk_bf   = (bf16*)(ws + alloc((size_t)DM * D3 * 2));
  bf16*  wv_bf   = (bf16*)(ws + alloc((size_t)DM * D3 * 2));
  bf16*  wfc_bf  = (bf16*)(ws + alloc((size_t)DM * DM * 2));
  bf16*  wg_bf   = (bf16*)(ws + alloc((size_t)DM * VD * 2));
  float* xm      = (float*)(ws + alloc((size_t)BATCH * MN * D3 * 4));
  float* possum  = (float*)(ws + alloc((size_t)MN * D3 * 4));
  bf16*  xvpe    = (bf16*)(ws + alloc((size_t)BATCH * MN * D3 * 2));
  float* gamma   = (float*)(ws + alloc((size_t)BATCH * MN * DM * 4));
  bf16*  Qbf     = (bf16*)(ws + alloc((size_t)BATCH * LSEQ * DM * 2));
  bf16*  Kbf     = (bf16*)(ws + alloc((size_t)BATCH * MN * DM * 2));
  bf16*  Vbf     = (bf16*)(ws + alloc((size_t)BATCH * MN * DM * 2));
  bf16*  attnbf  = (bf16*)(ws + alloc((size_t)BATCH * LSEQ * DM * 2));
  float* fcout   = (float*)(ws + alloc((size_t)BATCH * LSEQ * DM * 4));

  // 1) weight conversions
  auto cvt = [&](const float* s, bf16* d, int n) {
    cvt_f32_bf16_kernel<<<(n + 255) / 256, 256, 0, stream>>>(s, d, n);
  };
  cvt(Wq,   wq_bf,  DM * SD);
  cvt(Wk,   wk_bf,  DM * D3);
  cvt(Wv,   wv_bf,  DM * D3);
  cvt(fc_w, wfc_bf, DM * DM);
  cvt(Wg,   wg_bf,  DM * VD);

  // 2) velocity preprocessing
  msconv_kernel<<<BATCH * MN, 192, 0, stream>>>(x_velocity, ms_w3, ms_b3,
                                                ms_w5, ms_b5, ms_w7, ms_b7, xm);
  possum_kernel<<<MN, 192, 0, stream>>>(rel_emb, possum);
  relpe_kernel<<<BATCH * MN, 192, 0, stream>>>(xm, rel_emb, dw_w, dw_b, possum, xvpe);

  // 3) projections (WMMA GEMMs)
  // Gamma = sigmoid(x_velocity @ Wg^T)       [16384 x 256], K=64, f32 out
  gemm_wmma_kernel<VD, true, false, true><<<dim3(DM / 64, (BATCH * MN) / 128),
      256, 0, stream>>>(x_velocity, wg_bf, nullptr, gamma, DM, 1.0f);
  // Q = (x_spatial @ Wq^T) / sqrt(dk)        [32768 x 256], K=256, bf16 out
  gemm_wmma_kernel<SD, true, true, false><<<dim3(DM / 64, (BATCH * LSEQ) / 128),
      256, 0, stream>>>(x_spatial, wq_bf, nullptr, Qbf, DM, 0.1767766953f);
  // K = x_v_pe @ Wk^T                        [16384 x 256], K=192, bf16 out
  gemm_wmma_kernel<D3, false, true, false><<<dim3(DM / 64, (BATCH * MN) / 128),
      256, 0, stream>>>(xvpe, wk_bf, nullptr, Kbf, DM, 1.0f);
  // V = x_v_pe @ Wv^T
  gemm_wmma_kernel<D3, false, true, false><<<dim3(DM / 64, (BATCH * MN) / 128),
      256, 0, stream>>>(xvpe, wv_bf, nullptr, Vbf, DM, 1.0f);

  // 4) attention (flash-style, WMMA)
  attn_kernel<<<dim3(LSEQ / 128, NH, BATCH), 256, 0, stream>>>(Qbf, Kbf, Vbf, attnbf);

  // 5) fc projection (WMMA, fused bias, f32 out)
  gemm_wmma_kernel<DM, false, false, false><<<dim3(DM / 64, (BATCH * LSEQ) / 128),
      256, 0, stream>>>(attnbf, wfc_bf, fc_b, fcout, DM, 1.0f);

  // 6) gating + residual + LayerNorm
  ln_final_kernel<<<BATCH * LSEQ, 256, 0, stream>>>(fcout, gamma, x_spatial,
                                                    ln_w, ln_b, out);
}
